// SimplifiedLinearAttention_34162169872471
// MI455X (gfx1250) — matrix operand order repaired and checked by reading, not compiled, not run on hardware
//
#include <hip/hip_runtime.h>

// ---------------------------------------------------------------------------
// SimplifiedLinearAttention for MI455X (gfx1250), bf16 WMMA pipeline.
//   K1: qkv GEMM  [65536,384]x[384,1152] + bias/pos_enc/relu -> q/k/v ws (bf16)
//   K2: per (b,h): kv = k^T v (WMMA), z, q@kv (WMMA), + 5x5 depthwise conv
//   K3: proj GEMM [65536,384]x[384,384] + bias -> fp32 out
// ---------------------------------------------------------------------------

namespace {
constexpr int kDim   = 384;
constexpr int kHeads = 12;
constexpr int kHd    = 32;
constexpr int kNtok  = 256;
constexpr float kEps = 1e-6f;
}  // namespace

typedef __attribute__((ext_vector_type(8)))  __bf16 v8bf;
typedef __attribute__((ext_vector_type(16))) __bf16 v16bf;
typedef __attribute__((ext_vector_type(8)))  float  v8f;
typedef __attribute__((ext_vector_type(4)))  float  v4f;

__device__ __forceinline__ v8f zero8() {
  v8f z;
#pragma unroll
  for (int e = 0; e < 8; ++e) z[e] = 0.f;
  return z;
}

// A-matrix 16x32 bf16 fragment: lane m = lane%16, half = lane/16,
// elements 0..15 -> K = half*16 + e (contiguous, matches B loads).
__device__ __forceinline__ v16bf load_a_frag(const __bf16* rowbase, int half) {
  v8bf lo = *(const v8bf*)(rowbase + half * 16);
  v8bf hi = *(const v8bf*)(rowbase + half * 16 + 8);
  v16bf r;
#pragma unroll
  for (int e = 0; e < 8; ++e) { r[e] = lo[e]; r[e + 8] = hi[e]; }
  return r;
}

__device__ __forceinline__ v8f wmma_bf16(v16bf a, v16bf b, v8f c) {
  return __builtin_amdgcn_wmma_f32_16x16x32_bf16(false, a, false, b, (short)0, c,
                                                 false, false);
}

// ---------------------------------------------------------------------------
// Kernel 1: qkv GEMM.  grid=(512,9), block=256.  Tile 128(M)x128(N), K-step 32.
// B operand comes from qkv_w rows: B[k][n] = qkv_w[n][k] -> [N][K] contiguous.
// ---------------------------------------------------------------------------
__global__ __launch_bounds__(256) void sla_qkv_gemm(
    const float* __restrict__ x, const float* __restrict__ qkv_w,
    const float* __restrict__ qkv_b, const float* __restrict__ pos_enc,
    unsigned short* __restrict__ qws_, unsigned short* __restrict__ kws_,
    unsigned short* __restrict__ vws_) {
  __shared__ alignas(32) __bf16 sA[128 * 32];
  __shared__ alignas(32) __bf16 sB[128 * 32];
  const int tid  = threadIdx.x;
  const int lane = tid & 31, wave = tid >> 5;
  const int wm = wave >> 2, wn = wave & 3;  // 2x4 wave grid: 64 rows x 32 cols
  const int m0 = blockIdx.x * 128, n0 = blockIdx.y * 128;

  v8f acc[4][2];
#pragma unroll
  for (int mi = 0; mi < 4; ++mi)
#pragma unroll
    for (int ni = 0; ni < 2; ++ni) acc[mi][ni] = zero8();

  const int row = tid >> 1, kp = (tid & 1) * 16;  // 2 threads per tile row
  const float* asrc = x + (size_t)(m0 + row) * kDim + kp;
  const float* bsrc = qkv_w + (size_t)(n0 + row) * kDim + kp;

  for (int ks = 0; ks < kDim / 32; ++ks) {
    const int k0 = ks * 32;
    if (ks) __syncthreads();
#pragma unroll
    for (int q4 = 0; q4 < 4; ++q4) {
      v4f fa = *(const v4f*)(asrc + k0 + q4 * 4);
      v4f fb = *(const v4f*)(bsrc + k0 + q4 * 4);
      __bf16* da = &sA[row * 32 + kp + q4 * 4];
      __bf16* db = &sB[row * 32 + kp + q4 * 4];
      da[0] = (__bf16)fa.x; da[1] = (__bf16)fa.y; da[2] = (__bf16)fa.z; da[3] = (__bf16)fa.w;
      db[0] = (__bf16)fb.x; db[1] = (__bf16)fb.y; db[2] = (__bf16)fb.z; db[3] = (__bf16)fb.w;
    }
    if (ks + 1 < kDim / 32) {
      __builtin_prefetch(asrc + k0 + 32, 0, 1);
      __builtin_prefetch(bsrc + k0 + 32, 0, 1);
    }
    __syncthreads();

    v16bf af[4], bfr[2];
#pragma unroll
    for (int mi = 0; mi < 4; ++mi)
      af[mi] = load_a_frag(&sA[(wm * 64 + mi * 16 + (lane & 15)) * 32], lane >> 4);
#pragma unroll
    for (int ni = 0; ni < 2; ++ni)
      bfr[ni] = *(const v16bf*)&sB[(wn * 32 + ni * 16 + (lane & 15)) * 32 +
                                   (lane >> 4) * 16];
#pragma unroll
    for (int mi = 0; mi < 4; ++mi)
#pragma unroll
      for (int ni = 0; ni < 2; ++ni)
        acc[mi][ni] = wmma_bf16(af[mi], bfr[ni], acc[mi][ni]);
  }

  // Epilogue: bias, (pos_enc+)relu, scatter to head-major [BH][256][32] bf16.
  const int sec = n0 / kDim;  // 0=q, 1=k, 2=v (128 | 384 so tiles never span)
  __bf16* wsb = (__bf16*)(sec == 0 ? qws_ : (sec == 1 ? kws_ : vws_));
#pragma unroll
  for (int ni = 0; ni < 2; ++ni) {
    const int col  = n0 + wn * 32 + ni * 16 + (lane & 15);
    const float bias = qkv_b[col];
    const int cin = col - sec * kDim;
    const int h = cin >> 5, c = cin & 31;
#pragma unroll
    for (int mi = 0; mi < 4; ++mi) {
#pragma unroll
      for (int r = 0; r < 8; ++r) {
        const int m = m0 + wm * 64 + mi * 16 + r + ((lane >> 4) << 3);
        const int btok = m >> 8, ntok = m & 255;
        float v = acc[mi][ni][r] + bias;
        if (sec == 0)      v = fmaxf(v, 0.f);
        else if (sec == 1) v = fmaxf(v + pos_enc[ntok * kDim + cin], 0.f);
        wsb[((size_t)(btok * kHeads + h) * kNtok + ntok) * kHd + c] = (__bf16)v;
      }
    }
  }
}

// ---------------------------------------------------------------------------
// Kernel 2: per-(b,h) linear attention + depthwise conv.  grid=3072, block=256.
// ---------------------------------------------------------------------------
__global__ __launch_bounds__(256) void sla_attn(
    const unsigned short* __restrict__ qws_, const unsigned short* __restrict__ kws_,
    const unsigned short* __restrict__ vws_, const float* __restrict__ dwc_w,
    const float* __restrict__ dwc_b, unsigned short* __restrict__ hws_) {
  __shared__ alignas(32) __bf16 kT[32 * 256];   // k transposed [c][n]
  __shared__ alignas(32) __bf16 vT[32 * 256];   // v transposed [d][n]
  __shared__ alignas(32) __bf16 kvT[32 * 32];   // kv transposed [d][c]
  __shared__ __bf16 cv[256 * 32];               // conv result [n][c]
  __shared__ float  part[8 * 32];
  __shared__ float  ksum[32];
  __shared__ float  zz[256];

  const int bh  = blockIdx.x;
  const int tid = threadIdx.x, lane = tid & 31, wave = tid >> 5;
  const __bf16* qg = (const __bf16*)qws_ + (size_t)bh * (kNtok * kHd);
  const __bf16* kg = (const __bf16*)kws_ + (size_t)bh * (kNtok * kHd);
  const __bf16* vg = (const __bf16*)vws_ + (size_t)bh * (kNtok * kHd);

  // Stage k^T and v^T into LDS (transposed scatter).
#pragma unroll
  for (int s = 0; s < 4; ++s) {
    const int seg = s * 256 + tid;  // 1024 segments of 8 bf16
    const int n = seg >> 2, kp2 = (seg & 3) * 8;
    v8bf k8 = *(const v8bf*)(kg + n * 32 + kp2);
    v8bf v8 = *(const v8bf*)(vg + n * 32 + kp2);
#pragma unroll
    for (int e = 0; e < 8; ++e) {
      kT[(kp2 + e) * 256 + n] = k8[e];
      vT[(kp2 + e) * 256 + n] = v8[e];
    }
  }
  __syncthreads();

  // ksum[c] = sum_n k[n][c]
  {
    const int g = tid >> 5, c = tid & 31;
    const __bf16* kr = &kT[c * 256 + g * 32];
    float s = 0.f;
#pragma unroll
    for (int j = 0; j < 32; ++j) s += (float)kr[j];
    part[g * 32 + c] = s;
  }
  __syncthreads();
  if (tid < 32) {
    float s = 0.f;
#pragma unroll
    for (int g = 0; g < 8; ++g) s += part[g * 32 + tid];
    ksum[tid] = s;
  }
  __syncthreads();
  {  // z[n] = 1 / (q[n]·ksum + eps); q streamed straight from workspace
    const __bf16* qr = qg + tid * 32;
    float s = 0.f;
#pragma unroll
    for (int c = 0; c < 32; ++c) s += (float)qr[c] * ksum[c];
    zz[tid] = 1.f / (s + kEps);
  }
  __syncthreads();

  if (wave < 4) {
    // kv[c][d] = sum_j k[j][c] v[j][d]: one 16x16 tile per wave, K=256 in 8 steps
    const int ct = wave >> 1, dt = wave & 1;
    v8f acc = zero8();
#pragma unroll
    for (int js = 0; js < 8; ++js) {
      const int jb = js * 32;
      v16bf ak = load_a_frag(&kT[(ct * 16 + (lane & 15)) * 256 + jb], lane >> 4);
      v16bf bv = *(const v16bf*)&vT[(dt * 16 + (lane & 15)) * 256 + jb +
                                    (lane >> 4) * 16];
      acc = wmma_bf16(ak, bv, acc);
    }
#pragma unroll
    for (int r = 0; r < 8; ++r) {
      const int c = ct * 16 + r + ((lane >> 4) << 3);
      const int d = dt * 16 + (lane & 15);
      kvT[d * 32 + c] = (__bf16)acc[r];
    }
  } else {
    // Depthwise 5x5 conv over v (16x16 image per channel), 8 channels per wave.
    const int w4 = wave - 4;
#pragma unroll
    for (int ci = 0; ci < 8; ++ci) {
      const int c = w4 * 8 + ci;
      float wgt[25];
#pragma unroll
      for (int i = 0; i < 25; ++i) wgt[i] = dwc_w[c * 25 + i];
      const float bias = dwc_b[c];
      const __bf16* img = &vT[c * 256];
#pragma unroll
      for (int rep = 0; rep < 8; ++rep) {
        const int p = rep * 32 + lane;
        const int a = p >> 4, b = p & 15;
        float s = bias;
#pragma unroll
        for (int ky = 0; ky < 5; ++ky) {
          const int aa = a + ky - 2;
          if (aa < 0 || aa > 15) continue;
#pragma unroll
          for (int kx = 0; kx < 5; ++kx) {
            const int bb = b + kx - 2;
            if (bb < 0 || bb > 15) continue;
            s += (float)img[aa * 16 + bb] * wgt[ky * 5 + kx];
          }
        }
        cv[p * 32 + c] = (__bf16)s;
      }
    }
  }
  __syncthreads();

  // out[i][d] = z[i]*(q@kv)[i][d] + conv[i][d]; write merged [B][N][384] bf16.
  const int btok = bh / kHeads, h = bh % kHeads;
  __bf16* hout = (__bf16*)hws_;
#pragma unroll
  for (int t2 = 0; t2 < 2; ++t2) {
    const int rt = wave * 2 + t2;  // 16 row tiles over 8 waves
    v16bf aq = load_a_frag(qg + (rt * 16 + (lane & 15)) * 32, lane >> 4);
#pragma unroll
    for (int dt = 0; dt < 2; ++dt) {
      v16bf bk = *(const v16bf*)&kvT[(dt * 16 + (lane & 15)) * 32 +
                                     (lane >> 4) * 16];
      v8f acc = wmma_bf16(aq, bk, zero8());
#pragma unroll
      for (int r = 0; r < 8; ++r) {
        const int i = rt * 16 + r + ((lane >> 4) << 3);
        const int d = dt * 16 + (lane & 15);
        const float v = acc[r] * zz[i] + (float)cv[i * 32 + d];
        hout[((size_t)(btok * kNtok + i)) * kDim + h * kHd + d] = (__bf16)v;
      }
    }
  }
}

// ---------------------------------------------------------------------------
// Kernel 3: proj GEMM.  grid=(512,3), block=256.  Same tiling as kernel 1.
// ---------------------------------------------------------------------------
__global__ __launch_bounds__(256) void sla_proj_gemm(
    const unsigned short* __restrict__ hws_, const float* __restrict__ proj_w,
    const float* __restrict__ proj_b, float* __restrict__ out) {
  __shared__ alignas(32) __bf16 sA[128 * 32];
  __shared__ alignas(32) __bf16 sB[128 * 32];
  const __bf16* hws = (const __bf16*)hws_;
  const int tid  = threadIdx.x;
  const int lane = tid & 31, wave = tid >> 5;
  const int wm = wave >> 2, wn = wave & 3;
  const int m0 = blockIdx.x * 128, n0 = blockIdx.y * 128;

  v8f acc[4][2];
#pragma unroll
  for (int mi = 0; mi < 4; ++mi)
#pragma unroll
    for (int ni = 0; ni < 2; ++ni) acc[mi][ni] = zero8();

  const int row = tid >> 1, kp = (tid & 1) * 16;
  const __bf16* asrc = hws + (size_t)(m0 + row) * kDim + kp;
  const float* bsrc = proj_w + (size_t)(n0 + row) * kDim + kp;

  for (int ks = 0; ks < kDim / 32; ++ks) {
    const int k0 = ks * 32;
    if (ks) __syncthreads();
    *(v8bf*)&sA[row * 32 + kp]     = *(const v8bf*)(asrc + k0);
    *(v8bf*)&sA[row * 32 + kp + 8] = *(const v8bf*)(asrc + k0 + 8);
#pragma unroll
    for (int q4 = 0; q4 < 4; ++q4) {
      v4f fb = *(const v4f*)(bsrc + k0 + q4 * 4);
      __bf16* db = &sB[row * 32 + kp + q4 * 4];
      db[0] = (__bf16)fb.x; db[1] = (__bf16)fb.y; db[2] = (__bf16)fb.z; db[3] = (__bf16)fb.w;
    }
    if (ks + 1 < kDim / 32) {
      __builtin_prefetch(asrc + k0 + 32, 0, 1);
      __builtin_prefetch(bsrc + k0 + 32, 0, 1);
    }
    __syncthreads();

    v16bf af[4], bfr[2];
#pragma unroll
    for (int mi = 0; mi < 4; ++mi)
      af[mi] = load_a_frag(&sA[(wm * 64 + mi * 16 + (lane & 15)) * 32], lane >> 4);
#pragma unroll
    for (int ni = 0; ni < 2; ++ni)
      bfr[ni] = *(const v16bf*)&sB[(wn * 32 + ni * 16 + (lane & 15)) * 32 +
                                   (lane >> 4) * 16];
#pragma unroll
    for (int mi = 0; mi < 4; ++mi)
#pragma unroll
      for (int ni = 0; ni < 2; ++ni)
        acc[mi][ni] = wmma_bf16(af[mi], bfr[ni], acc[mi][ni]);
  }

#pragma unroll
  for (int ni = 0; ni < 2; ++ni) {
    const int col = n0 + wn * 32 + ni * 16 + (lane & 15);
    const float bias = proj_b[col];
#pragma unroll
    for (int mi = 0; mi < 4; ++mi) {
#pragma unroll
      for (int r = 0; r < 8; ++r) {
        const int m = m0 + wm * 64 + mi * 16 + r + ((lane >> 4) << 3);
        out[(size_t)m * kDim + col] = acc[mi][ni][r] + bias;
      }
    }
  }
}

// ---------------------------------------------------------------------------
extern "C" void kernel_launch(void* const* d_in, const int* in_sizes, int n_in,
                              void* d_out, int out_size, void* d_ws, size_t ws_size,
                              hipStream_t stream) {
  (void)in_sizes; (void)n_in; (void)out_size; (void)ws_size;
  const float* x      = (const float*)d_in[0];
  const float* qkv_w  = (const float*)d_in[1];
  const float* qkv_b  = (const float*)d_in[2];
  const float* pos    = (const float*)d_in[3];
  const float* dwc_w  = (const float*)d_in[4];
  const float* dwc_b  = (const float*)d_in[5];
  const float* proj_w = (const float*)d_in[6];
  const float* proj_b = (const float*)d_in[7];
  float* out = (float*)d_out;

  // Workspace: q/k/v head-major [3072][256][32] bf16 + merged head out, bf16.
  const size_t kHSZ = (size_t)3072 * 256 * 32;  // elements per tensor
  unsigned short* qws = (unsigned short*)d_ws;
  unsigned short* kws = qws + kHSZ;
  unsigned short* vws = kws + kHSZ;
  unsigned short* hws = vws + kHSZ;

  sla_qkv_gemm<<<dim3(512, 9), dim3(256), 0, stream>>>(x, qkv_w, qkv_b, pos,
                                                       qws, kws, vws);
  sla_attn<<<dim3(3072), dim3(256), 0, stream>>>(qws, kws, vws, dwc_w, dwc_b, hws);
  sla_proj_gemm<<<dim3(512, 3), dim3(256), 0, stream>>>(hws, proj_w, proj_b, out);
}
